// GAT_57200374448520
// MI455X (gfx1250) — matrix operand order repaired and checked by reading, not compiled, not run on hardware
//
#include <hip/hip_runtime.h>
#include <hip/hip_bf16.h>
#include <cstdint>

// ---------------- problem constants (match reference) ----------------
#define GAT_N   50000
#define GAT_E   800000
#define GAT_ET  (GAT_N + GAT_E)   // self-loops appended
#define GAT_K   128               // input channels to every GEMM (compile-time)
#define NEG_SLOPE 0.2f

typedef __attribute__((ext_vector_type(16))) __bf16        v16bf;
typedef __attribute__((ext_vector_type(8)))  float         v8f;
typedef __attribute__((ext_vector_type(4)))  unsigned int  v4u;

// ---------------- prep: W[K x Ncol] (f32, row-major) -> Wt[Ncol x K] (bf16, packed)
__global__ void gat_prep_wt(const float* __restrict__ W, unsigned short* __restrict__ Wt,
                            int Ncol) {
  int i = blockIdx.x * blockDim.x + threadIdx.x;     // over Ncol*K
  if (i >= GAT_K * Ncol) return;
  int col = i / GAT_K, k = i % GAT_K;
  __bf16 h = (__bf16)W[(size_t)k * Ncol + col];      // native v_cvt RNE
  Wt[i] = __builtin_bit_cast(unsigned short, h);
}

// ---------------- GEMM: C[M,Ncol] = A[M,128] * W. One 16-row block per wave32.
// A fragments (all 4 K-steps) held in registers and reused across all N-tiles.
__global__ void gat_wmma_gemm(const float* __restrict__ A,
                              const unsigned short* __restrict__ Wt,
                              float* __restrict__ C, int Ncol, int ntn, int mtiles) {
  int mt = blockIdx.x * blockDim.y + threadIdx.y;
  if (mt >= mtiles) return;                 // uniform per-wave -> EXEC all-1s inside
  int lane = threadIdx.x;                   // wave32 lane
  int half = lane >> 4, mr = lane & 15;
  const float* Arow = A + (size_t)(mt * 16 + mr) * GAT_K;

  // A frag (contiguous k order, same as B): lane half 0 -> K = k0+0..15
  //                               lane half 1 -> K = k0+16..31
  v16bf afr[4];
#pragma unroll
  for (int s = 0; s < 4; ++s) {
    int ka = s * 32 + 16 * half;
#pragma unroll
    for (int i = 0; i < 8; ++i) afr[s][i]     = (__bf16)Arow[ka + i];
#pragma unroll
    for (int i = 0; i < 8; ++i) afr[s][8 + i] = (__bf16)Arow[ka + 8 + i];
  }

  for (int nt = 0; nt < ntn; ++nt) {
    int colB = nt * 16 + mr;
    const unsigned short* wcol = Wt + (size_t)colB * GAT_K;
    v8f acc = {};
#pragma unroll
    for (int s = 0; s < 4; ++s) {
      // B frag: lanes 0-15 hold K = k0+0..15 of col n, lanes 16-31 hold K = k0+16..31
      // -> 16 contiguous bf16 from the transposed weight copy (32B, 16B-aligned)
      union { v16bf v; v4u q[2]; } b;
      const v4u* wp = (const v4u*)(wcol + s * 32 + 16 * half);
      b.q[0] = wp[0];
      b.q[1] = wp[1];
      acc = __builtin_amdgcn_wmma_f32_16x16x32_bf16(false, afr[s], false, b.v,
                                                    (short)0, acc, false, false);
    }
    // C frag: VGPR r -> row mt*16 + r + 8*half, col nt*16 + mr
    float* Cp = C + (size_t)(mt * 16 + 8 * half) * Ncol + nt * 16 + mr;
#pragma unroll
    for (int r = 0; r < 8; ++r) Cp[(size_t)r * Ncol] = acc[r];
  }
}

// ---------------- per-node attention coefficients: a_src/a_dst [N, heads]
__global__ void gat_att_coef(const float* __restrict__ H, const float* __restrict__ as_,
                             const float* __restrict__ ad_, float* __restrict__ aS,
                             float* __restrict__ aD, int heads, int chead) {
  int i = blockIdx.x * blockDim.x + threadIdx.x;
  if (i >= GAT_N * heads) return;
  int n = i / heads, h = i % heads;
  const float* hp = H + (size_t)n * heads * chead + (size_t)h * chead;
  float s1 = 0.f, s2 = 0.f;
  for (int c = 0; c < chead; ++c) {
    float v = hp[c];
    s1 += v * as_[h * chead + c];
    s2 += v * ad_[h * chead + c];
  }
  aS[i] = s1; aD[i] = s2;
}

__global__ void gat_fill(float* __restrict__ p, float v, int n) {
  int i = blockIdx.x * blockDim.x + threadIdx.x;
  if (i < n) p[i] = v;
}

__device__ __forceinline__ long long esrc(const long long* ei, int e) {
  return (e < GAT_E) ? ei[e] : (long long)(e - GAT_E);
}
__device__ __forceinline__ long long edst(const long long* ei, int e) {
  return (e < GAT_E) ? ei[(size_t)GAT_E + e] : (long long)(e - GAT_E);
}

// ---------------- edge pass 1: leaky-relu logits + segment max (atomic, int trick)
__global__ void gat_edge_logits(const long long* __restrict__ ei,
                                const float* __restrict__ aS, const float* __restrict__ aD,
                                float* __restrict__ eTmp, float* __restrict__ mx, int heads) {
  int i = blockIdx.x * blockDim.x + threadIdx.x;
  if (i >= GAT_ET * heads) return;
  int e = i / heads, h = i % heads;
  long long s = esrc(ei, e), d = edst(ei, e);
  float v = aS[s * heads + h] + aD[d * heads + h];
  v = (v > 0.f) ? v : NEG_SLOPE * v;
  eTmp[i] = v;
  float* addr = &mx[d * heads + h];
  if (v >= 0.f) atomicMax((int*)addr, __float_as_int(v));
  else          atomicMin((unsigned int*)addr, __float_as_uint(v));
}

// ---------------- edge pass 2: exp(e - max) + segment sum
__global__ void gat_edge_exp(const long long* __restrict__ ei, float* __restrict__ eTmp,
                             const float* __restrict__ mx, float* __restrict__ dn, int heads) {
  int i = blockIdx.x * blockDim.x + threadIdx.x;
  if (i >= GAT_ET * heads) return;
  int e = i / heads, h = i % heads;
  long long d = edst(ei, e);
  float ex = expf(eTmp[i] - mx[d * heads + h]);
  eTmp[i] = ex;
  atomicAdd(&dn[d * heads + h], ex);
}

// ---------------- edge pass 3: normalized message scatter-add
__global__ void gat_edge_message(const long long* __restrict__ ei,
                                 const float* __restrict__ eTmp, const float* __restrict__ dn,
                                 const float* __restrict__ H, float* __restrict__ out,
                                 int heads, int chead) {
  int i = blockIdx.x * blockDim.x + threadIdx.x;
  if (i >= GAT_ET * heads) return;
  int e = i / heads, h = i % heads;
  long long s = esrc(ei, e), d = edst(ei, e);
  float alpha = eTmp[i] / (dn[d * heads + h] + 1e-16f);
  const float* hs = H + (size_t)s * heads * chead + (size_t)h * chead;
  float* od = out + (size_t)d * heads * chead + (size_t)h * chead;
  for (int c = 0; c < chead; ++c) atomicAdd(&od[c], alpha * hs[c]);
}

// ---------------- bias (+ optional ELU)
__global__ void gat_bias_act(const float* __restrict__ in, const float* __restrict__ bias,
                             float* __restrict__ out, int total, int cols, int do_elu) {
  int i = blockIdx.x * blockDim.x + threadIdx.x;
  if (i >= total) return;
  float v = in[i] + bias[i % cols];
  if (do_elu) v = (v > 0.f) ? v : (expf(v) - 1.f);
  out[i] = v;
}

// ---------------- host-side layer driver ----------------
static void run_gat_layer(hipStream_t st, const float* Xin, const float* W,
                          const float* as_, const float* ad_, const float* bias,
                          int heads, int chead, int do_elu,
                          float* H, float* OUT, float* aS, float* aD,
                          float* MX, float* DN, float* ETMP, unsigned short* WT,
                          const long long* ei, float* dst) {
  const int Ncol = heads * chead;
  const int ntn = Ncol / 16;
  const int mtiles = GAT_N / 16;

  int nw = GAT_K * Ncol;
  gat_prep_wt<<<(nw + 255) / 256, 256, 0, st>>>(W, WT, Ncol);

  dim3 blk(32, 4);
  gat_wmma_gemm<<<(mtiles + 3) / 4, blk, 0, st>>>(Xin, WT, H, Ncol, ntn, mtiles);

  int nh = GAT_N * heads;
  gat_att_coef<<<(nh + 255) / 256, 256, 0, st>>>(H, as_, ad_, aS, aD, heads, chead);

  const float ninf = __builtin_bit_cast(float, 0xFF800000u);
  gat_fill<<<(nh + 255) / 256, 256, 0, st>>>(MX, ninf, nh);
  gat_fill<<<(nh + 255) / 256, 256, 0, st>>>(DN, 0.f, nh);
  int nout = GAT_N * Ncol;
  gat_fill<<<(nout + 255) / 256, 256, 0, st>>>(OUT, 0.f, nout);

  int eh = GAT_ET * heads;
  gat_edge_logits<<<(eh + 255) / 256, 256, 0, st>>>(ei, aS, aD, ETMP, MX, heads);
  gat_edge_exp<<<(eh + 255) / 256, 256, 0, st>>>(ei, ETMP, MX, DN, heads);
  gat_edge_message<<<(eh + 255) / 256, 256, 0, st>>>(ei, ETMP, DN, H, OUT, heads, chead);

  gat_bias_act<<<(nout + 255) / 256, 256, 0, st>>>(OUT, bias, dst, nout, Ncol, do_elu);
}

extern "C" void kernel_launch(void* const* d_in, const int* in_sizes, int n_in,
                              void* d_out, int out_size, void* d_ws, size_t ws_size,
                              hipStream_t stream) {
  const float*     x   = (const float*)d_in[0];
  const long long* ei  = (const long long*)d_in[1];   // int64 [2, E]
  const float* W1 = (const float*)d_in[2];
  const float* as1 = (const float*)d_in[3];
  const float* ad1 = (const float*)d_in[4];
  const float* b1  = (const float*)d_in[5];
  const float* W2 = (const float*)d_in[6];
  const float* as2 = (const float*)d_in[7];
  const float* ad2 = (const float*)d_in[8];
  const float* b2  = (const float*)d_in[9];
  const float* W3 = (const float*)d_in[10];
  const float* as3 = (const float*)d_in[11];
  const float* ad3 = (const float*)d_in[12];
  const float* b3  = (const float*)d_in[13];
  float* out = (float*)d_out;

  // workspace carve-up (floats): H, OUT, X2 : N*128 each; aS/aD/MX/DN : N*8;
  // ETMP : ET*8; WT : 128*128 bf16 (stored as ushort, 16B-aligned)
  float* ws   = (float*)d_ws;
  float* H    = ws;
  float* OUT  = H   + (size_t)GAT_N * 128;
  float* X2   = OUT + (size_t)GAT_N * 128;
  float* aS   = X2  + (size_t)GAT_N * 128;
  float* aD   = aS  + (size_t)GAT_N * 8;
  float* MX   = aD  + (size_t)GAT_N * 8;
  float* DN   = MX  + (size_t)GAT_N * 8;
  float* ETMP = DN  + (size_t)GAT_N * 8;
  unsigned short* WT = (unsigned short*)(ETMP + (size_t)GAT_ET * 8);

  // layer 1: 128 -> 8 heads x 16, concat, ELU  (result in X2)
  run_gat_layer(stream, x, W1, as1, ad1, b1, 8, 16, 1,
                H, OUT, aS, aD, MX, DN, ETMP, WT, ei, X2);
  // layer 2: 128 -> 8 heads x 16, concat, ELU  (reads X2, overwrites X2 after consumption)
  run_gat_layer(stream, X2, W2, as2, ad2, b2, 8, 16, 1,
                H, OUT, aS, aD, MX, DN, ETMP, WT, ei, X2);
  // layer 3: 128 -> 1 head x 32, no concat (mean over 1 head == identity), no ELU
  run_gat_layer(stream, X2, W3, as3, ad3, b3, 1, 32, 0,
                H, OUT, aS, aD, MX, DN, ETMP, WT, ei, out);
}
